// RoPEMultiheadAttention_19172734009945
// MI455X (gfx1250) — matrix operand order repaired and checked by reading, not compiled, not run on hardware
//
#include <hip/hip_runtime.h>

#define EDIM 1024
#define SEQ  2048
#define BATCH 2
#define NH   16
#define HD   64
#define LDA  40   // 32 k-halves + 8 pad  (80B stride, 16B aligned)
#define LDK  72   // 64 halves + 8 pad    (144B stride, 16B aligned)

typedef __attribute__((ext_vector_type(16))) _Float16 v16h;
typedef __attribute__((ext_vector_type(8)))  _Float16 v8h;
typedef __attribute__((ext_vector_type(4)))  _Float16 v4h;
typedef __attribute__((ext_vector_type(8)))  float    v8f;

typedef int v4i __attribute__((vector_size(16)));
typedef __attribute__((address_space(1))) v4i gv4i;
typedef __attribute__((address_space(3))) v4i lv4i;

// 16-byte async DMA copy global -> LDS (ASYNCcnt-tracked, no VGPR roundtrip)
__device__ __forceinline__ void acp16B(const _Float16* g, _Float16* l) {
  __builtin_amdgcn_global_load_async_to_lds_b128(
      (gv4i*)const_cast<_Float16*>(g), (lv4i*)l, 0, 0);
}

__device__ __forceinline__ v16h mk16(v8h lo, v8h hi) {
  v16h r;
#pragma unroll
  for (int i = 0; i < 8; i++) { r[i] = lo[i]; r[i + 8] = hi[i]; }
  return r;
}

__device__ __forceinline__ v8f wmma_f16(v16h a, v16h b, v8f c) {
  return __builtin_amdgcn_wmma_f32_16x16x32_f16(false, a, false, b, (short)0, c,
                                                false, false);
}

// A-matrix fragment (16x32 f16): lanes 0-15 row M=lane, K=k0+[0..7],[16..23];
// lanes 16-31 same rows, K=k0+[8..15],[24..31].
__device__ __forceinline__ v16h frag_a(const _Float16* base, int ld, int k0) {
  int lane = threadIdx.x & 31;
  const _Float16* p = base + (lane & 15) * ld + k0 + ((lane >> 4) << 3);
  v8h lo = *(const v8h*)p;
  v8h hi = *(const v8h*)(p + 16);
  return mk16(lo, hi);
}

// B-matrix fragment (32x16 f16), stored column-major [n][k]:
// lanes 0-15 col N=lane K=k0+[0..7],[16..23]; lanes 16-31 col N=lane-16 K=k0+[8..15],[24..31].
__device__ __forceinline__ v16h frag_b(const _Float16* base, int ld, int k0) {
  int lane = threadIdx.x & 31;
  const _Float16* p = base + (lane & 15) * ld + k0 + ((lane >> 4) << 3);
  v8h lo = *(const v8h*)p;
  v8h hi = *(const v8h*)(p + 16);
  return mk16(lo, hi);
}

// xor-exchange within 16-lane rows via v_permlane16_b32 (VALU, not DS pipe)
__device__ __forceinline__ float permrow(float x, unsigned s0, unsigned s1) {
  return __int_as_float(__builtin_amdgcn_permlane16(
      __float_as_int(x), __float_as_int(x), (int)s0, (int)s1, false, false));
}
__device__ __forceinline__ float permrow_x1(float x) {  // xor 1
  return permrow(x, 0x67452301u, 0xEFCDAB89u);
}
__device__ __forceinline__ float row_max16(float x) {
  x = fmaxf(x, permrow(x, 0xFEDCBA98u, 0x76543210u));  // xor 8
  x = fmaxf(x, permrow(x, 0x32107654u, 0xBA98FEDCu));  // xor 4
  x = fmaxf(x, permrow(x, 0x54761032u, 0xDCFE98BAu));  // xor 2
  x = fmaxf(x, permrow_x1(x));                          // xor 1
  return x;
}
__device__ __forceinline__ float row_sum16(float x) {
  x += permrow(x, 0xFEDCBA98u, 0x76543210u);
  x += permrow(x, 0x32107654u, 0xBA98FEDCu);
  x += permrow(x, 0x54761032u, 0xDCFE98BAu);
  x += permrow_x1(x);
  return x;
}

// ---------------------------------------------------------------- convert
__global__ __launch_bounds__(256) void cvt_f32_f16(const float* __restrict__ in,
                                                   _Float16* __restrict__ out,
                                                   int n4) {
  int i = blockIdx.x * 256 + threadIdx.x;
  if (i < n4) {
    float4 v = ((const float4*)in)[i];
    v4h h;
    h[0] = (_Float16)v.x; h[1] = (_Float16)v.y;
    h[2] = (_Float16)v.z; h[3] = (_Float16)v.w;
    ((v4h*)out)[i] = h;
  }
}

// ------------------------------------------------------------- GEMM (f16)
// C[M=4096,N=1024] = X[M,K] @ W^T + bias.
// MODE 0: f16 out [B,H,S,D] + RoPE;  1: f16 out [B,H,S,D];  2: f32 out [M,E].
template <int MODE>
__global__ __launch_bounds__(256) void gemm_kernel(
    const _Float16* __restrict__ X, const _Float16* __restrict__ Wg,
    const float* __restrict__ bias, void* __restrict__ outv) {
  __shared__ _Float16 Al[2][128 * LDA];
  __shared__ _Float16 Bl[2][128 * LDA];
  int m_blk = blockIdx.x, n_blk = blockIdx.y;
  int t = threadIdx.x;
  int w = t >> 5, lane = t & 31;
  int wm = (w & 1) * 64, wn = (w >> 1) * 32;
  int lrow = t >> 1, lkq = (t & 1) * 16;
  const _Float16* ag = X  + (size_t)(m_blk * 128 + lrow) * EDIM + lkq;
  const _Float16* bg = Wg + (size_t)(n_blk * 128 + lrow) * EDIM + lkq;

  v8f acc[4][2] = {};
  const int NK = EDIM / 32;

  // prologue: async-stage K-step 0 into buffer 0
  {
    _Float16* ad = &Al[0][lrow * LDA + lkq];
    _Float16* bd = &Bl[0][lrow * LDA + lkq];
    acp16B(ag, ad);      acp16B(ag + 8, ad + 8);
    acp16B(bg, bd);      acp16B(bg + 8, bd + 8);
  }

  for (int k = 0; k < NK; k++) {
    int buf = k & 1;
    __builtin_amdgcn_s_wait_asynccnt(0);  // this wave's copies for buf done
    __syncthreads();                      // everyone's copies done + prev reads done
    if (k + 1 < NK) {                     // stage next K-step into other buffer
      const _Float16* ag2 = ag + (k + 1) * 32;
      const _Float16* bg2 = bg + (k + 1) * 32;
      _Float16* ad = &Al[buf ^ 1][lrow * LDA + lkq];
      _Float16* bd = &Bl[buf ^ 1][lrow * LDA + lkq];
      acp16B(ag2, ad);     acp16B(ag2 + 8, ad + 8);
      acp16B(bg2, bd);     acp16B(bg2 + 8, bd + 8);
    }
#pragma unroll
    for (int mt = 0; mt < 4; mt++) {
      v16h af = frag_a(&Al[buf][(wm + mt * 16) * LDA], LDA, 0);
#pragma unroll
      for (int nt = 0; nt < 2; nt++) {
        v16h bf = frag_b(&Bl[buf][(wn + nt * 16) * LDA], LDA, 0);
        acc[mt][nt] = wmma_f16(af, bf, acc[mt][nt]);
      }
    }
  }

  int nbase = n_blk * 128 + wn;
  int mbase = m_blk * 128 + wm;
  int rh = (lane >> 4) << 3;
  int cl = lane & 15;
#pragma unroll
  for (int mt = 0; mt < 4; mt++) {
#pragma unroll
    for (int nt = 0; nt < 2; nt++) {
      int N = nbase + nt * 16 + cl;
      float bv = bias[N];
      if (MODE == 2) {
        float* out = (float*)outv;
#pragma unroll
        for (int v = 0; v < 8; v++) {
          int M = mbase + mt * 16 + rh + v;
          out[(size_t)M * EDIM + N] = acc[mt][nt][v] + bv;
        }
      } else {
        _Float16* out = (_Float16*)outv;
        int hN = N >> 6;
        int dN = N & 63;
        float inv = exp2f((float)(dN >> 1) * (-13.287712379549449f / 32.0f));
#pragma unroll
        for (int v = 0; v < 8; v++) {
          int M = mbase + mt * 16 + rh + v;
          float val = acc[mt][nt][v] + bv;
          if (MODE == 0) {  // RoPE: pair (even,odd) head dims live in lanes (2j,2j+1)
            int spos = M & (SEQ - 1);
            float cc, ss;
            __sincosf((float)spos * inv, &ss, &cc);
            float other = permrow_x1(val);
            val = (dN & 1) ? (val * cc + other * ss) : (val * cc - other * ss);
          }
          int b = M >> 11;
          int sp = M & (SEQ - 1);
          out[(((size_t)b * NH + hN) * SEQ + sp) * HD + dN] = (_Float16)val;
        }
      }
    }
  }
}

// ---------------------------------------------------------- flash attention
// grid (S/128, B*H); each of 8 waves owns 16 query rows.
__global__ __launch_bounds__(256) void flash_kernel(
    const _Float16* __restrict__ Q, const _Float16* __restrict__ K,
    const _Float16* __restrict__ V, _Float16* __restrict__ O) {
  __shared__ _Float16 Kl[64 * LDK];        // [t][d] row-major  (async-staged)
  __shared__ _Float16 Vl[64 * LDK];        // [d][t] transposed (manual staging)
  __shared__ _Float16 Pl[8 * 16 * LDK];    // per-wave P staging
  int bh = blockIdx.y;
  const _Float16* Qh = Q + (size_t)bh * SEQ * HD;
  const _Float16* Kh = K + (size_t)bh * SEQ * HD;
  const _Float16* Vh = V + (size_t)bh * SEQ * HD;
  int w = threadIdx.x >> 5, lane = threadIdx.x & 31;
  int q0 = blockIdx.x * 128 + w * 16;
  int rh = (lane >> 4) << 3, cl = lane & 15;

  // Q fragments stay in registers; fold 1/sqrt(D)=0.125 in here (exact in f16)
  v16h qa[2];
#pragma unroll
  for (int ks = 0; ks < 2; ks++) {
    const _Float16* p = Qh + (size_t)(q0 + cl) * HD + ks * 32 + ((lane >> 4) << 3);
    qa[ks] = mk16(*(const v8h*)p, *(const v8h*)(p + 16));
    qa[ks] = qa[ks] * (_Float16)0.125f;
  }

  v8f o[4] = {};
  float mrow[8], lrow[8];
#pragma unroll
  for (int v = 0; v < 8; v++) { mrow[v] = -1e30f; lrow[v] = 0.f; }
  _Float16* Pw = Pl + w * 16 * LDK;
  int srow = threadIdx.x >> 2;
  int sdq = (threadIdx.x & 3) * 16;

  for (int kb = 0; kb < SEQ / 64; kb++) {
    {  // stage K asynchronously, V manually transposed
      const _Float16* kp = Kh + (size_t)(kb * 64 + srow) * HD + sdq;
      _Float16* kd = &Kl[srow * LDK + sdq];
      acp16B(kp, kd);
      acp16B(kp + 8, kd + 8);
      const _Float16* vp = Vh + (size_t)(kb * 64 + srow) * HD + sdq;
      v8h v0 = *(const v8h*)vp, v1 = *(const v8h*)(vp + 8);
#pragma unroll
      for (int i = 0; i < 8; i++) {
        Vl[(sdq + i) * LDK + srow]     = v0[i];
        Vl[(sdq + 8 + i) * LDK + srow] = v1[i];
      }
    }
    __builtin_amdgcn_s_wait_asynccnt(0);
    __syncthreads();

    v8f sa[4] = {};
#pragma unroll
    for (int nt = 0; nt < 4; nt++)
#pragma unroll
      for (int ks = 0; ks < 2; ks++) {
        v16h kf = frag_b(&Kl[nt * 16 * LDK], LDK, ks * 32);
        sa[nt] = wmma_f16(qa[ks], kf, sa[nt]);
      }

    // online softmax; row reductions via v_permlane16 (VALU, not DS pipe)
#pragma unroll
    for (int v = 0; v < 8; v++) {
      float mx = fmaxf(fmaxf(sa[0][v], sa[1][v]), fmaxf(sa[2][v], sa[3][v]));
      mx = row_max16(mx);
      float mn = fmaxf(mrow[v], mx);
      float sc = __expf(mrow[v] - mn);
      mrow[v] = mn;
      float rs = 0.f;
#pragma unroll
      for (int nt = 0; nt < 4; nt++) {
        float p = __expf(sa[nt][v] - mn);
        sa[nt][v] = p;
        rs += p;
      }
      rs = row_sum16(rs);
      lrow[v] = lrow[v] * sc + rs;
#pragma unroll
      for (int nt = 0; nt < 4; nt++) o[nt][v] *= sc;
    }

    // C-layout -> A-layout via wave-private LDS (in-order DS, no barrier)
#pragma unroll
    for (int nt = 0; nt < 4; nt++)
#pragma unroll
      for (int v = 0; v < 8; v++)
        Pw[(rh + v) * LDK + nt * 16 + cl] = (_Float16)sa[nt][v];

    v16h pa[2];
#pragma unroll
    for (int ks = 0; ks < 2; ks++) pa[ks] = frag_a(Pw, LDK, ks * 32);
#pragma unroll
    for (int nt = 0; nt < 4; nt++)
#pragma unroll
      for (int ks = 0; ks < 2; ks++) {
        v16h vb = frag_b(&Vl[nt * 16 * LDK], LDK, ks * 32);
        o[nt] = wmma_f16(pa[ks], vb, o[nt]);
      }
    __syncthreads();
  }

  int b = bh >> 4, h = bh & 15;
#pragma unroll
  for (int v = 0; v < 8; v++) {
    float inv = 1.f / lrow[v];
    int s = q0 + rh + v;
#pragma unroll
    for (int nt = 0; nt < 4; nt++) {
      int d = nt * 16 + cl;
      O[(((size_t)b * SEQ + s) * NH + h) * HD + d] = (_Float16)(o[nt][v] * inv);
    }
  }
}

// ------------------------------------------------------------------ launcher
extern "C" void kernel_launch(void* const* d_in, const int* in_sizes, int n_in,
                              void* d_out, int out_size, void* d_ws,
                              size_t ws_size, hipStream_t stream) {
  (void)in_sizes; (void)n_in; (void)out_size; (void)ws_size;
  const float* q_in = (const float*)d_in[0];
  const float* k_in = (const float*)d_in[1];
  const float* v_in = (const float*)d_in[2];
  const float* Wq   = (const float*)d_in[3];
  const float* bq   = (const float*)d_in[4];
  const float* Wk   = (const float*)d_in[5];
  const float* bk   = (const float*)d_in[6];
  const float* Wv   = (const float*)d_in[7];
  const float* bv   = (const float*)d_in[8];
  const float* Wo   = (const float*)d_in[9];
  const float* bo   = (const float*)d_in[10];
  float* out = (float*)d_out;

  _Float16* ws = (_Float16*)d_ws;
  const size_t EE = (size_t)EDIM * EDIM;           // 1M halves
  const size_t TE = (size_t)BATCH * SEQ * EDIM;    // 4M halves
  _Float16* Wq16 = ws;
  _Float16* Wk16 = ws + EE;
  _Float16* Wv16 = ws + 2 * EE;
  _Float16* Wo16 = ws + 3 * EE;
  _Float16* Xq16 = ws + 4 * EE;
  _Float16* Xk16 = Xq16 + TE;
  _Float16* Xv16 = Xk16 + TE;
  _Float16* Q16  = Xv16 + TE;
  _Float16* K16  = Q16 + TE;
  _Float16* V16  = K16 + TE;
  _Float16* A16  = Xq16;  // reuse: Xq16 fully consumed before flash runs

  int nw4 = (int)(EE / 4);
  dim3 cw((nw4 + 255) / 256);
  cvt_f32_f16<<<cw, 256, 0, stream>>>(Wq, Wq16, nw4);
  cvt_f32_f16<<<cw, 256, 0, stream>>>(Wk, Wk16, nw4);
  cvt_f32_f16<<<cw, 256, 0, stream>>>(Wv, Wv16, nw4);
  cvt_f32_f16<<<cw, 256, 0, stream>>>(Wo, Wo16, nw4);
  int na4 = (int)(TE / 4);
  dim3 ca((na4 + 255) / 256);
  cvt_f32_f16<<<ca, 256, 0, stream>>>(q_in, Xq16, na4);
  cvt_f32_f16<<<ca, 256, 0, stream>>>(k_in, Xk16, na4);
  cvt_f32_f16<<<ca, 256, 0, stream>>>(v_in, Xv16, na4);

  dim3 pg(BATCH * SEQ / 128, EDIM / 128);  // (32, 8)
  gemm_kernel<0><<<pg, 256, 0, stream>>>(Xq16, Wq16, bq, (void*)Q16);
  gemm_kernel<0><<<pg, 256, 0, stream>>>(Xk16, Wk16, bk, (void*)K16);
  gemm_kernel<1><<<pg, 256, 0, stream>>>(Xv16, Wv16, bv, (void*)V16);

  dim3 fg(SEQ / 128, BATCH * NH);  // (16, 32)
  flash_kernel<<<fg, 256, 0, stream>>>(Q16, K16, V16, A16);

  gemm_kernel<2><<<pg, 256, 0, stream>>>(A16, Wo16, bo, (void*)out);
}
